// SwinTransformerBlock_4612794876530
// MI455X (gfx1250) — compile-verified, matrix operand order checked by reading, not run on hardware
//
#include <hip/hip_runtime.h>
#include <hip/hip_bf16.h>
#include <cstddef>
#include <cstdint>

// ---------------------------------------------------------------------------
// Swin Transformer block for MI455X (gfx1250), wave32, WMMA f16->f32,
// async global->LDS staging (ASYNCcnt).
// B=64 H=W=56 C=96 nh=3 d=32 WS=7 SS=3 N=49 nW=64 B_=4096 M=200704
// ---------------------------------------------------------------------------

typedef __attribute__((ext_vector_type(16))) _Float16 v16h;
typedef __attribute__((ext_vector_type(8)))  _Float16 v8h;
typedef __attribute__((ext_vector_type(8)))  float    v8f;

#define M_TOT   200704
#define C_DIM   96
#define QKV_DIM 288
#define HID_DIM 384

// workspace layout (bytes) — all offsets 32B-aligned
static constexpr size_t SZ_A   = (size_t)M_TOT * C_DIM   * 2;   // 38,535,168
static constexpr size_t SZ_QKV = (size_t)M_TOT * QKV_DIM * 2;   // 115,605,504
static constexpr size_t OFF_A  = 0;                // xw, later LN2 acts (f16)
static constexpr size_t OFF_B  = SZ_A;             // qkv; later h [M,384] f16
static constexpr size_t OFF_C  = OFF_B + SZ_QKV;   // attn_out [M,96] f16
static constexpr size_t OFF_W  = OFF_C + SZ_A;     // packed f16 weights
static constexpr size_t SZ_WQ  = (size_t)QKV_DIM * C_DIM * 2;   // 55,296
static constexpr size_t SZ_WP  = (size_t)C_DIM * C_DIM * 2;     // 18,432
static constexpr size_t SZ_W1  = (size_t)HID_DIM * C_DIM * 2;   // 73,728
static constexpr size_t SZ_W2  = (size_t)C_DIM * HID_DIM * 2;   // 73,728
static constexpr size_t OFF_BM = OFF_W + SZ_WQ + SZ_WP + SZ_W1 + SZ_W2; // bias+mask f32

__device__ __forceinline__ v8f wmma_f16(v16h a, v16h b, v8f c) {
  return __builtin_amdgcn_wmma_f32_16x16x32_f16(false, a, false, b, (short)0, c,
                                                false, false);
}

// Async DMA: 16B per enabled lane, global -> LDS, tracked by ASYNCcnt.
__device__ __forceinline__ void async_b128(void* lds, const void* gptr) {
  uint32_t loff = (uint32_t)(uintptr_t)lds;   // aperture low bits = LDS offset
  asm volatile("global_load_async_to_lds_b128 %0, %1, off"
               :: "v"(loff), "v"(gptr) : "memory");
}
__device__ __forceinline__ void wait_async0() {
  asm volatile("s_wait_asynccnt 0x0" ::: "memory");
}

// A fragment (16x32 f16) from LDS, ISA wave32 layout: lanes 0-15 -> rows,
// K halves {0..7,16..23}; lanes 16-31 -> same rows, K halves {8..15,24..31}.
// ld must be a multiple of 8 halves so both chunks are 16B-aligned b128 loads.
__device__ __forceinline__ v16h frag_a(const _Float16* s, int ld, int row0, int k0) {
  int lane = threadIdx.x & 31;
  int r  = row0 + (lane & 15);
  int ks = k0 + ((lane >> 4) << 3);
  v16h f;
#pragma unroll
  for (int i = 0; i < 8; ++i) f[i]     = s[r * ld + ks + i];
#pragma unroll
  for (int i = 0; i < 8; ++i) f[i + 8] = s[r * ld + ks + 16 + i];
  return f;
}

// B fragment straight from packed weights in global memory (L2-resident):
// layout [kb][nt][lane][16 halves] -> one contiguous 32B read per lane.
__device__ __forceinline__ v16h frag_w(const _Float16* __restrict__ wp, int NT,
                                       int kb, int nt) {
  int lane = threadIdx.x & 31;
  return *(const v16h*)(wp + (((size_t)kb * NT + nt) * 32 + lane) * 16);
}

// ---------------------------------------------------------------------------
// K0a: pack weight W[N,K] f32 into WMMA-B fragment-major f16 layout.
// ---------------------------------------------------------------------------
__global__ void wt_pack(const float* __restrict__ w, _Float16* __restrict__ wp,
                        int N, int K) {
  int idx = blockIdx.x * 256 + threadIdx.x;
  int NT = N / 16, KB = K / 32;
  if (idx >= KB * NT * 32) return;
  int lane = idx & 31, frag = idx >> 5;
  int nt = frag % NT, kb = frag / NT;
  int c  = nt * 16 + (lane & 15);
  int hs = (lane >> 4) << 3;
  _Float16* dst = wp + (size_t)idx * 16;
#pragma unroll
  for (int i = 0; i < 8; ++i) dst[i]     = (_Float16)w[(size_t)c * K + kb * 32 + hs + i];
#pragma unroll
  for (int i = 0; i < 8; ++i) dst[8 + i] = (_Float16)w[(size_t)c * K + kb * 32 + 16 + hs + i];
}

// K0b: fuse rel-pos bias gather + attention mask -> bm[wm][h][nq][nk] f32
__global__ void bias_prep(const float* __restrict__ tbl,
                          const int* __restrict__ relidx,
                          const float* __restrict__ amask,
                          float* __restrict__ bm) {
  int i = blockIdx.x * 256 + threadIdx.x;   // over 64*3*2401
  if (i < 64 * 3 * 2401) {
    int n = i % 2401, h = (i / 2401) % 3, wm = i / (3 * 2401);
    bm[i] = tbl[relidx[n] * 3 + h] + amask[(size_t)wm * 2401 + n];
  }
}

// ---------------------------------------------------------------------------
// K1: LN1 + cyclic shift(-3,-3) + window partition. One wave32 per row.
// ---------------------------------------------------------------------------
__global__ __launch_bounds__(256) void ln1_shift_part(
    const float* __restrict__ x, const float* __restrict__ g,
    const float* __restrict__ b, _Float16* __restrict__ xw) {
  int row  = blockIdx.x * 8 + (threadIdx.x >> 5);
  int lane = threadIdx.x & 31;
  int n = row % 49, win = row / 49;
  int wx = win & 7, wy = (win >> 3) & 7, bb = win >> 6;
  int iy = n / 7, ix = n % 7;
  int hh = (wy * 7 + iy + 3) % 56;   // xs[i] = xn[(i+3) mod 56]
  int ww = (wx * 7 + ix + 3) % 56;
  size_t base = ((size_t)bb * 3136 + hh * 56 + ww) * C_DIM;
  float v0 = x[base + lane], v1 = x[base + lane + 32], v2 = x[base + lane + 64];
  float s = v0 + v1 + v2, ss = v0 * v0 + v1 * v1 + v2 * v2;
#pragma unroll
  for (int o = 16; o > 0; o >>= 1) {
    s  += __shfl_xor(s, o, 32);
    ss += __shfl_xor(ss, o, 32);
  }
  float mean = s * (1.f / 96.f);
  float var  = ss * (1.f / 96.f) - mean * mean;
  float inv  = rsqrtf(var + 1e-5f);
  size_t ob = (size_t)row * C_DIM;
  xw[ob + lane]      = (_Float16)((v0 - mean) * inv * g[lane]      + b[lane]);
  xw[ob + lane + 32] = (_Float16)((v1 - mean) * inv * g[lane + 32] + b[lane + 32]);
  xw[ob + lane + 64] = (_Float16)((v2 - mean) * inv * g[lane + 64] + b[lane + 64]);
}

// ---------------------------------------------------------------------------
// Generic WMMA GEMM. Out[M,N] = A[M,K] @ W^T + bias. Tile 128x96, 4 waves,
// each wave 64x48 = 12 WMMAs per 32-deep k-step. A staged to LDS with
// global_load_async_to_lds_b128; B fragments from packed global (L2).
//   epi 0: f16 store   epi 1: GELU f16 store   epi 2: f32 + residual store
// ---------------------------------------------------------------------------
__global__ __launch_bounds__(128) void gemm_wmma(
    const _Float16* __restrict__ A, const _Float16* __restrict__ Wp,
    const float* __restrict__ bias, void* __restrict__ Out,
    const float* __restrict__ Res, int M, int N, int K, int NT, int epi) {
  __shared__ _Float16 As[128][40];
  int m_base = blockIdx.x * 128;
  int n_base = blockIdx.y * 96;
  int wave = threadIdx.x >> 5;
  int m0w = (wave & 1) * 64;
  int nt_base = blockIdx.y * 6 + (wave >> 1) * 3;
  v8f acc[4][3];
#pragma unroll
  for (int mi = 0; mi < 4; ++mi)
#pragma unroll
    for (int ni = 0; ni < 3; ++ni)
#pragma unroll
      for (int e = 0; e < 8; ++e) acc[mi][ni][e] = 0.f;

  int KB = K >> 5;
  for (int kb = 0; kb < KB; ++kb) {
    int k0 = kb << 5;
#pragma unroll
    for (int it = 0; it < 4; ++it) {           // A tile 128x32 -> LDS (async)
      int chunk = threadIdx.x + it * 128;      // 512 chunks of 8 halves
      int r = chunk >> 2, co = (chunk & 3) << 3;
      async_b128(&As[r][co], A + (size_t)(m_base + r) * K + k0 + co);
    }
    if (kb + 1 < KB)                           // prefetch next A slice
      __builtin_prefetch(A + (size_t)(m_base + threadIdx.x) * K + k0 + 32, 0, 1);
    wait_async0();
    __syncthreads();
    v16h af[4];
#pragma unroll
    for (int mi = 0; mi < 4; ++mi)
      af[mi] = frag_a(&As[0][0], 40, m0w + mi * 16, 0);
#pragma unroll
    for (int ni = 0; ni < 3; ++ni) {
      v16h bf = frag_w(Wp, NT, kb, nt_base + ni);
#pragma unroll
      for (int mi = 0; mi < 4; ++mi)
        acc[mi][ni] = wmma_f16(af[mi], bf, acc[mi][ni]);
    }
    __syncthreads();
  }

  int lane = threadIdx.x & 31;
  int colp = lane & 15, rsel = (lane >> 4) << 3;
  int n0w = (wave >> 1) * 48;
#pragma unroll
  for (int mi = 0; mi < 4; ++mi)
#pragma unroll
    for (int ni = 0; ni < 3; ++ni)
#pragma unroll
      for (int r = 0; r < 8; ++r) {
        int gm = m_base + m0w + mi * 16 + rsel + r;
        int gn = n_base + n0w + ni * 16 + colp;
        float v = acc[mi][ni][r] + bias[gn];
        size_t idx = (size_t)gm * N + gn;
        if (epi == 0) {
          ((_Float16*)Out)[idx] = (_Float16)v;
        } else if (epi == 1) {
          float gelu = 0.5f * v * (1.f + erff(v * 0.70710678118654752f));
          ((_Float16*)Out)[idx] = (_Float16)gelu;
        } else {
          ((float*)Out)[idx] = v + Res[idx];
        }
      }
}

// ---------------------------------------------------------------------------
// K3: windowed attention. Block = (head, window), 128 threads (4 waves).
// K tile via async DMA (pad rows zero-filled, disjoint addresses). Q scaled
// in VGPRs; V stored transposed so the P@V B-fragment reads contiguously.
// ---------------------------------------------------------------------------
__global__ __launch_bounds__(128) void attn_kernel(
    const _Float16* __restrict__ qkv, const float* __restrict__ bm,
    _Float16* __restrict__ aout) {
  int h   = blockIdx.x;   // 0..2
  int win = blockIdx.y;   // 0..4095
  __shared__ _Float16 Qs[64][40], Ks[64][40];
  __shared__ _Float16 VsT[32][72];             // [dd][n'] transposed
  __shared__ float    Sf[64][67];
  __shared__ _Float16 Ps[64][72];
  const size_t rowbase = (size_t)win * 49;

  // K: async global->LDS for valid rows, zero-fill pads (disjoint)
#pragma unroll
  for (int it = 0; it < 2; ++it) {
    int chunk = threadIdx.x + it * 128;
    int r = chunk >> 2, co = (chunk & 3) << 3;
    if (r < 49) {
      async_b128(&Ks[r][co], qkv + (rowbase + r) * QKV_DIM + 96 + h * 32 + co);
    } else {
#pragma unroll
      for (int i = 0; i < 8; ++i) Ks[r][co + i] = (_Float16)0.f;
    }
  }
  // Q (scaled) and V (transposed) through VGPRs
#pragma unroll
  for (int it = 0; it < 2; ++it) {
    int chunk = threadIdx.x + it * 128;
    int r = chunk >> 2, co = (chunk & 3) << 3;
    if (r < 49) {
      const _Float16* q = qkv + (rowbase + r) * QKV_DIM + h * 32 + co;
#pragma unroll
      for (int i = 0; i < 8; ++i)
        Qs[r][co + i] = (_Float16)((float)q[i] * 0.17677669529663687f);
      v8h val = *(const v8h*)(qkv + (rowbase + r) * QKV_DIM + 192 + h * 32 + co);
#pragma unroll
      for (int i = 0; i < 8; ++i) VsT[co + i][r] = val[i];
    } else {
#pragma unroll
      for (int i = 0; i < 8; ++i) Qs[r][co + i] = (_Float16)0.f;
#pragma unroll
      for (int i = 0; i < 8; ++i) VsT[co + i][r] = (_Float16)0.f;
    }
  }
  wait_async0();
  __syncthreads();

  int wave = threadIdx.x >> 5;
  int lane = threadIdx.x & 31;
  int colp = lane & 15, rsel = (lane >> 4) << 3;
  int mt = wave;          // each wave owns a 16-row stripe
  const float* bmh = bm + ((size_t)(win & 63) * 3 + h) * 2401;

  v16h aQ = frag_a(&Qs[0][0], 40, mt * 16, 0);
#pragma unroll
  for (int nt = 0; nt < 4; ++nt) {
    v16h bK = frag_a(&Ks[0][0], 40, nt * 16, 0);  // rows of K == cols of K^T
    v8f sc;
#pragma unroll
    for (int e = 0; e < 8; ++e) sc[e] = 0.f;
    sc = wmma_f16(aQ, bK, sc);
#pragma unroll
    for (int r = 0; r < 8; ++r) {
      int nq = mt * 16 + rsel + r;
      int nk = nt * 16 + colp;
      float v = sc[r];
      if (nk >= 49)
        v = -1e30f;
      else if (nq < 49)
        v += bmh[nq * 49 + nk];
      Sf[nq][nk] = v;
    }
  }
  __syncthreads();

  if (threadIdx.x < 64) {   // row softmax over 64 (padded) columns
    int r = threadIdx.x;
    float mx = -3.4e38f;
    for (int c = 0; c < 64; ++c) mx = fmaxf(mx, Sf[r][c]);
    float sum = 0.f;
    for (int c = 0; c < 64; ++c) {
      float e = __expf(Sf[r][c] - mx);
      Sf[r][c] = e;
      sum += e;
    }
    float inv = 1.f / sum;
    for (int c = 0; c < 64; ++c) Ps[r][c] = (_Float16)(Sf[r][c] * inv);
  }
  __syncthreads();

  v8f o[2];
#pragma unroll
  for (int ni = 0; ni < 2; ++ni)
#pragma unroll
    for (int e = 0; e < 8; ++e) o[ni][e] = 0.f;
#pragma unroll
  for (int kt = 0; kt < 2; ++kt) {
    v16h aP = frag_a(&Ps[0][0], 72, mt * 16, kt * 32);
#pragma unroll
    for (int ni = 0; ni < 2; ++ni) {
      v16h bV = frag_a(&VsT[0][0], 72, ni * 16, kt * 32);
      o[ni] = wmma_f16(aP, bV, o[ni]);
    }
  }
#pragma unroll
  for (int ni = 0; ni < 2; ++ni)
#pragma unroll
    for (int r = 0; r < 8; ++r) {
      int nq = mt * 16 + rsel + r;
      if (nq < 49)
        aout[(rowbase + nq) * C_DIM + h * 32 + ni * 16 + colp] =
            (_Float16)o[ni][r];
    }
}

// ---------------------------------------------------------------------------
// K4: proj GEMM + window reverse + un-shift(+3,+3) + residual + LN2.
// ---------------------------------------------------------------------------
__global__ __launch_bounds__(128) void proj_kernel(
    const _Float16* __restrict__ Aatt, const _Float16* __restrict__ Wp,
    const float* __restrict__ pb, const float* __restrict__ x,
    const float* __restrict__ g2, const float* __restrict__ b2,
    float* __restrict__ x1, _Float16* __restrict__ ln2h) {
  __shared__ _Float16 As[64][40];
  __shared__ float Ct[64][97];
  int m_base = blockIdx.x * 64;
  int wave = threadIdx.x >> 5;
  int m0w = (wave & 1) * 32;
  int nt_base = (wave >> 1) * 3;
  v8f acc[2][3];
#pragma unroll
  for (int mi = 0; mi < 2; ++mi)
#pragma unroll
    for (int ni = 0; ni < 3; ++ni)
#pragma unroll
      for (int e = 0; e < 8; ++e) acc[mi][ni][e] = 0.f;

  for (int kb = 0; kb < 3; ++kb) {
    int k0 = kb << 5;
#pragma unroll
    for (int it = 0; it < 2; ++it) {           // A tile 64x32 -> LDS (async)
      int chunk = threadIdx.x + it * 128;
      int r = chunk >> 2, co = (chunk & 3) << 3;
      async_b128(&As[r][co], Aatt + (size_t)(m_base + r) * 96 + k0 + co);
    }
    wait_async0();
    __syncthreads();
    v16h af0 = frag_a(&As[0][0], 40, m0w, 0);
    v16h af1 = frag_a(&As[0][0], 40, m0w + 16, 0);
#pragma unroll
    for (int ni = 0; ni < 3; ++ni) {
      v16h bf = frag_w(Wp, 6, kb, nt_base + ni);
      acc[0][ni] = wmma_f16(af0, bf, acc[0][ni]);
      acc[1][ni] = wmma_f16(af1, bf, acc[1][ni]);
    }
    __syncthreads();
  }

  int lane = threadIdx.x & 31;
  int colp = lane & 15, rsel = (lane >> 4) << 3;
  int n0w = (wave >> 1) * 48;
#pragma unroll
  for (int mi = 0; mi < 2; ++mi)
#pragma unroll
    for (int ni = 0; ni < 3; ++ni)
#pragma unroll
      for (int r = 0; r < 8; ++r)
        Ct[m0w + mi * 16 + rsel + r][n0w + ni * 16 + colp] = acc[mi][ni][r];
  __syncthreads();

  if (threadIdx.x < 64) {
    int r = threadIdx.x;
    int grow = m_base + r;
    int n = grow % 49, win = grow / 49;
    int wx = win & 7, wy = (win >> 3) & 7, bb = win >> 6;
    int iy = n / 7, ix = n % 7;
    int hh = (wy * 7 + iy + 3) % 56;   // reverse shift: +3
    int ww = (wx * 7 + ix + 3) % 56;
    size_t base = ((size_t)bb * 3136 + hh * 56 + ww) * C_DIM;
    float s = 0.f, ss = 0.f;
    for (int c = 0; c < 96; ++c) {
      float v = Ct[r][c] + pb[c] + x[base + c];
      x1[base + c] = v;
      s += v;
      ss += v * v;
    }
    float mean = s * (1.f / 96.f);
    float var  = ss * (1.f / 96.f) - mean * mean;
    float inv  = rsqrtf(var + 1e-5f);
    for (int c = 0; c < 96; ++c) {
      float v = x1[base + c];
      ln2h[base + c] = (_Float16)((v - mean) * inv * g2[c] + b2[c]);
    }
  }
}

// ---------------------------------------------------------------------------
extern "C" void kernel_launch(void* const* d_in, const int* in_sizes, int n_in,
                              void* d_out, int out_size, void* d_ws,
                              size_t ws_size, hipStream_t stream) {
  const float* x      = (const float*)d_in[0];
  const float* amask  = (const float*)d_in[1];
  const int*   relidx = (const int*)d_in[2];
  const float* n1g    = (const float*)d_in[3];
  const float* n1b    = (const float*)d_in[4];
  const float* qkv_w  = (const float*)d_in[5];
  const float* qkv_b  = (const float*)d_in[6];
  const float* tbl    = (const float*)d_in[7];
  const float* proj_w = (const float*)d_in[8];
  const float* proj_b = (const float*)d_in[9];
  const float* n2g    = (const float*)d_in[10];
  const float* n2b    = (const float*)d_in[11];
  const float* fc1_w  = (const float*)d_in[12];
  const float* fc1_b  = (const float*)d_in[13];
  const float* fc2_w  = (const float*)d_in[14];
  const float* fc2_b  = (const float*)d_in[15];
  float* out = (float*)d_out;

  char* ws = (char*)d_ws;
  _Float16* xw    = (_Float16*)(ws + OFF_A);
  _Float16* qkvb  = (_Float16*)(ws + OFF_B);
  _Float16* aoutp = (_Float16*)(ws + OFF_C);
  _Float16* hbuf  = (_Float16*)(ws + OFF_B);  // aliases qkv+aout, used after
  _Float16* wqkv  = (_Float16*)(ws + OFF_W);
  _Float16* wproj = (_Float16*)(ws + OFF_W + SZ_WQ);
  _Float16* wfc1  = (_Float16*)(ws + OFF_W + SZ_WQ + SZ_WP);
  _Float16* wfc2  = (_Float16*)(ws + OFF_W + SZ_WQ + SZ_WP + SZ_W1);
  float*    bmbuf = (float*)(ws + OFF_BM);

  wt_pack<<<(288 * 96 / 16 + 255) / 256, 256, 0, stream>>>(qkv_w, wqkv, 288, 96);
  wt_pack<<<(96 * 96 / 16 + 255) / 256, 256, 0, stream>>>(proj_w, wproj, 96, 96);
  wt_pack<<<(384 * 96 / 16 + 255) / 256, 256, 0, stream>>>(fc1_w, wfc1, 384, 96);
  wt_pack<<<(96 * 384 / 16 + 255) / 256, 256, 0, stream>>>(fc2_w, wfc2, 96, 384);
  bias_prep<<<(64 * 3 * 2401 + 255) / 256, 256, 0, stream>>>(tbl, relidx, amask, bmbuf);

  ln1_shift_part<<<M_TOT / 8, 256, 0, stream>>>(x, n1g, n1b, xw);

  // qkv: [M,96] @ [96,288]
  gemm_wmma<<<dim3(M_TOT / 128, 3), 128, 0, stream>>>(
      xw, wqkv, qkv_b, (void*)qkvb, nullptr, M_TOT, QKV_DIM, C_DIM, 18, 0);

  attn_kernel<<<dim3(3, 4096), 128, 0, stream>>>(qkvb, bmbuf, aoutp);

  // proj + scatter + residual + LN2 (x1 lives in d_out)
  proj_kernel<<<M_TOT / 64, 128, 0, stream>>>(aoutp, wproj, proj_b, x, n2g, n2b,
                                              out, xw);

  // fc1 + GELU: [M,96] @ [96,384]
  gemm_wmma<<<dim3(M_TOT / 128, 4), 128, 0, stream>>>(
      xw, wfc1, fc1_b, (void*)hbuf, nullptr, M_TOT, HID_DIM, C_DIM, 24, 1);

  // fc2 + residual: [M,384] @ [384,96] + x1 -> d_out
  gemm_wmma<<<dim3(M_TOT / 128, 1), 128, 0, stream>>>(
      hbuf, wfc2, fc2_b, (void*)out, out, M_TOT, C_DIM, HID_DIM, 6, 2);
}
